// AttentionPooling_66915590471715
// MI455X (gfx1250) — compile-verified, matrix operand order checked by reading, not run on hardware
//
#include <hip/hip_runtime.h>
#include <hip/hip_bf16.h>

typedef __bf16 bf16_t;
typedef __attribute__((ext_vector_type(16))) __bf16 v16bf;
typedef __attribute__((ext_vector_type(8)))  float  v8f;

#define DIM   128
#define T_LEN 200
#define TP    208          // T padded to multiple of 16
#define HSTR  136          // LDS row stride for x (bf16 elems) -> 272B, conflict-free
#define WSTR  136          // LDS row stride for W^T (bf16 elems)

// dynamic LDS layout (bytes):
//   x=hist+target bf16 [TP][HSTR]  : 208*136*2 = 56576
//   Wt bf16 [DIM][WSTR]            : 128*136*2 = 34816
//   q f32[128], bias f32[128]
//   logits/scores f32[256], red f32[16], part f32[256]
#define SMEM_BYTES (TP*HSTR*2 + DIM*WSTR*2 + 4*(DIM*2 + 256 + 16 + 256))

union Frag { v16bf v; unsigned int u[8]; };

__global__ __launch_bounds__(256)
void AttentionPooling_66915590471715_kernel(
        const float* __restrict__ target,   // [B,128]
        const float* __restrict__ hist,     // [B,200,128]
        const float* __restrict__ Wk,       // [128,128] row-major [k][n]
        const float* __restrict__ Wb,       // [128]
        const float* __restrict__ qk,       // [128]
        float* __restrict__ out)            // [B,128]
{
    extern __shared__ char smem[];
    bf16_t* s_x      = (bf16_t*)smem;                 // [TP][HSTR]  x = hist+target (bf16)
    bf16_t* s_wt     = s_x + TP*HSTR;                 // [DIM][WSTR] Wt[n][k] (bf16)
    float*  s_qk     = (float*)(s_wt + DIM*WSTR);     // [128]
    float*  s_bias   = s_qk + DIM;                    // [128]
    float*  s_logits = s_bias + DIM;                  // [256] logits -> scores
    float*  s_red    = s_logits + 256;                // [16]
    float*  s_part   = s_red + 16;                    // [256]

    const int tid  = threadIdx.x;
    const int b    = blockIdx.x;
    const int wave = tid >> 5;
    const int lane = tid & 31;
    const int half = lane >> 4;   // which 16-lane half of the wave
    const int l16  = lane & 15;

    const float* hg = hist + (size_t)b * T_LEN * DIM;

    // ---- stage W^T -> LDS bf16 (coalesced global read, transposed LDS write) ----
    for (int idx = tid; idx < DIM*DIM; idx += 256) {
        int k = idx >> 7, n = idx & 127;
        s_wt[n*WSTR + k] = (bf16_t)Wk[idx];
    }
    // ---- stage x = hist + target -> LDS bf16 (target value hoists: column
    //      index is constant per thread across the strided loop) ----
    {
        const int d0 = tid & 127;
        const int t0 = tid >> 7;
        const float tgv = target[(size_t)b*DIM + d0];
        for (int t = t0; t < T_LEN; t += 2)
            s_x[t*HSTR + d0] = (bf16_t)(hg[(size_t)t*DIM + d0] + tgv);
        for (int t = T_LEN + t0; t < TP; t += 2)     // zero pad rows [200,208)
            s_x[t*HSTR + d0] = (bf16_t)0.0f;
    }
    if (tid < DIM) {
        s_qk[tid]   = qk[tid];
        s_bias[tid] = Wb[tid];
    }
    __syncthreads();

    // ---- phase 1: per-wave M-tiles, logits via WMMA bf16 (f32 accum) ----
    for (int mt = wave; mt < TP/16; mt += 8) {
        const int row = mt*16 + l16;   // A-matrix row for this lane

        // A fragments: pure u32 LDS gathers in the ISA 7.12.2 16-bit layout:
        // elem e -> K = 32c + e + 8*(e>=8) + 8*half.
        Frag a[4];
        #pragma unroll
        for (int c = 0; c < 4; ++c) {
            #pragma unroll
            for (int j = 0; j < 8; ++j) {
                const int k = 32*c + 2*j + ((j >= 4) ? 8 : 0) + 8*half;
                a[c].u[j] = *(const unsigned int*)(&s_x[row*HSTR + k]);
            }
        }

        float part[8];
        #pragma unroll
        for (int r = 0; r < 8; ++r) part[r] = 0.0f;

        #pragma unroll 2
        for (int nt = 0; nt < 8; ++nt) {
            const int n = nt*16 + l16;   // B-matrix column for this lane
            Frag w[4];
            #pragma unroll
            for (int c = 0; c < 4; ++c) {
                #pragma unroll
                for (int j = 0; j < 8; ++j) {
                    const int k = 32*c + 2*j + ((j >= 4) ? 8 : 0) + 8*half;
                    w[c].u[j] = *(const unsigned int*)(&s_wt[n*WSTR + k]);
                }
            }
            v8f acc = {0.f,0.f,0.f,0.f,0.f,0.f,0.f,0.f};
            acc = __builtin_amdgcn_wmma_f32_16x16x32_bf16(false, a[0].v, false, w[0].v, (short)0, acc, false, false);
            acc = __builtin_amdgcn_wmma_f32_16x16x32_bf16(false, a[1].v, false, w[1].v, (short)0, acc, false, false);
            acc = __builtin_amdgcn_wmma_f32_16x16x32_bf16(false, a[2].v, false, w[2].v, (short)0, acc, false, false);
            acc = __builtin_amdgcn_wmma_f32_16x16x32_bf16(false, a[3].v, false, w[3].v, (short)0, acc, false, false);

            // h = relu(acc + bias[n]); logit partial += h * q[n]
            const float bn = s_bias[n];
            const float qn = s_qk[n];
            #pragma unroll
            for (int r = 0; r < 8; ++r) {
                float hv = fmaxf(acc[r] + bn, 0.0f);
                part[r] = fmaf(hv, qn, part[r]);
            }
        }

        // Reduce over the 16 lanes sharing each output row (C/D layout:
        // VGPR r, lanes 0-15 -> M=r, lanes 16-31 -> M=r+8).
        #pragma unroll
        for (int r = 0; r < 8; ++r) {
            float p = part[r];
            p += __shfl_xor(p, 1, 16);
            p += __shfl_xor(p, 2, 16);
            p += __shfl_xor(p, 4, 16);
            p += __shfl_xor(p, 8, 16);
            part[r] = p;
        }
        if (l16 == 0) {
            #pragma unroll
            for (int r = 0; r < 8; ++r)
                s_logits[mt*16 + 8*half + r] = part[r];
        }
    }
    __syncthreads();

    // ---- phase 2: softmax over T=200 (q_bias is shift-invariant, dropped) ----
    float lg = (tid < T_LEN) ? s_logits[tid] : -3.4e38f;
    float m = lg;
    m = fmaxf(m, __shfl_xor(m, 1, 32));
    m = fmaxf(m, __shfl_xor(m, 2, 32));
    m = fmaxf(m, __shfl_xor(m, 4, 32));
    m = fmaxf(m, __shfl_xor(m, 8, 32));
    m = fmaxf(m, __shfl_xor(m, 16, 32));
    if (lane == 0) s_red[wave] = m;
    __syncthreads();
    float mm = s_red[0];
    #pragma unroll
    for (int w = 1; w < 8; ++w) mm = fmaxf(mm, s_red[w]);
    __syncthreads();   // all reads of s_red done before reuse

    float ev = (tid < T_LEN) ? __expf(lg - mm) : 0.0f;
    s_logits[tid] = ev;          // scores
    float sv = ev;
    sv += __shfl_xor(sv, 1, 32);
    sv += __shfl_xor(sv, 2, 32);
    sv += __shfl_xor(sv, 4, 32);
    sv += __shfl_xor(sv, 8, 32);
    sv += __shfl_xor(sv, 16, 32);
    if (lane == 0) s_red[wave] = sv;
    __syncthreads();
    float tot = s_red[0];
    #pragma unroll
    for (int w = 1; w < 8; ++w) tot += s_red[w];
    const float inv = 1.0f / tot;

    // ---- phase 3: out = inv * sum_t score_t * hist_t (f32, L2-resident) ----
    const int dd = tid & 127;
    const int hh = tid >> 7;                 // split T into two halves of 100
    const float* hp = hg + (size_t)(hh*100)*DIM + dd;
    float acc = 0.0f;
    for (int t = 0; t < 100; ++t)
        acc = fmaf(s_logits[hh*100 + t], hp[(size_t)t*DIM], acc);
    s_part[tid] = acc;
    __syncthreads();
    if (tid < DIM)
        out[(size_t)b*DIM + tid] = inv * (s_part[tid] + s_part[tid + DIM]);
}

extern "C" void kernel_launch(void* const* d_in, const int* in_sizes, int n_in,
                              void* d_out, int out_size, void* d_ws, size_t ws_size,
                              hipStream_t stream) {
    const float* target = (const float*)d_in[0];
    const float* hist   = (const float*)d_in[1];
    const float* Wk     = (const float*)d_in[2];
    const float* Wb     = (const float*)d_in[3];
    const float* qk     = (const float*)d_in[4];
    // d_in[5] (q_bias) intentionally unused: softmax is invariant to a
    // constant shift of the logits.
    float* out = (float*)d_out;

    const int B = in_sizes[0] / DIM;   // 16384

    // >64KB dynamic LDS: raise the limit (no-op if already allowed).
    (void)hipFuncSetAttribute((const void*)AttentionPooling_66915590471715_kernel,
                              hipFuncAttributeMaxDynamicSharedMemorySize,
                              SMEM_BYTES);

    AttentionPooling_66915590471715_kernel<<<dim3(B), dim3(256), SMEM_BYTES, stream>>>(
        target, hist, Wk, Wb, qk, out);
}
